// MultitaskEarlyRanker_32435593019420
// MI455X (gfx1250) — compile-verified, matrix operand order checked by reading, not run on hardware
//
#include <hip/hip_runtime.h>
#include <hip/hip_bf16.h>

// ---------------------------------------------------------------------------
// MultitaskEarlyRanker on gfx1250 (CDNA5, wave32, WMMA + async-to-LDS)
//   B=16384, IU=256, H=1024, E=8, D=128, T=8
// Block = 256 threads (8 waves), 32 rows (two 16-row M-panels), 1024 cols.
// Each B fragment feeds 2 WMMAs (one per M-panel) -> halved B traffic and
// halved exposed-load waits per WMMA vs 16-row blocking.
// ---------------------------------------------------------------------------

#define BB   16384
#define IU   256
#define HH   1024
#define EE   8
#define DD   128
#define ED   1024   // E*D
#define TT   8
#define PADE 8      // f16 elements of row padding in LDS A panel (16 bytes)
#define RPB  32     // rows per block

typedef __attribute__((ext_vector_type(16))) _Float16 v16h;
typedef __attribute__((ext_vector_type(8)))  _Float16 v8h;
typedef __attribute__((ext_vector_type(8)))  float    v8f;

// ---------------------------------------------------------------------------
// Async 16-byte global -> LDS copy (per-lane), tracked by ASYNCcnt.
// Generic-pointer low 32 bits == LDS byte offset (ISA §10.2 aperture map).
// ---------------------------------------------------------------------------
__device__ inline void async_cp16(void* lds_dst, const void* gsrc) {
    unsigned off = (unsigned)(unsigned long long)(uintptr_t)lds_dst;
    asm volatile("global_load_async_to_lds_b128 %0, %1, off"
                 :: "v"(off), "v"(gsrc) : "memory");
}
__device__ inline void wait_async0() {
    asm volatile("s_wait_asynccnt 0x0" ::: "memory");
}

// ---------------------------------------------------------------------------
// A fragment from an LDS panel laid out [rows][K + PADE] f16.
// ISA 16-bit A 16x32 layout: lane<16 row=lane K={kb..kb+7,kb+16..kb+23} kb=0,
// lanes 16-31 row=lane-16, kb=8  ->  two contiguous 16B chunks per lane.
// ---------------------------------------------------------------------------
__device__ inline v16h lds_a_frag(const _Float16* s_a, int rowstride,
                                  int kbase, int lane) {
    const _Float16* p = s_a + (lane & 15) * rowstride
                            + kbase + ((lane >> 4) << 3);
    v16h r;
    v8h* h = reinterpret_cast<v8h*>(&r);
    h[0] = *reinterpret_cast<const v8h*>(p);
    h[1] = *reinterpret_cast<const v8h*>(p + 16);
    return r;
}

__device__ inline v8f wmma_f16(v16h a, v16h b, v8f c) {
    return __builtin_amdgcn_wmma_f32_16x16x32_f16(
        false, a, false, b, (short)0, c, false, false);
}

// ---------------------------------------------------------------------------
// fp32 -> f16 row-major convert (user_features)
// ---------------------------------------------------------------------------
__global__ void cvt_f32_to_f16(const float* __restrict__ src,
                               _Float16* __restrict__ dst, int n) {
    for (int i = blockIdx.x * blockDim.x + threadIdx.x; i < n;
         i += gridDim.x * blockDim.x) {
        dst[i] = (_Float16)src[i];
    }
}

// ---------------------------------------------------------------------------
// Pack row-major fp32 [K x N] weight into lane-contiguous B-fragment layout:
//   dst[((kt*(N/16)+nt)*32 + lane)*16 + i]
//   k = kt*32 + ((lane>>4)<<3) + (i<8 ? i : i+8),  n = nt*16 + (lane&15)
// ---------------------------------------------------------------------------
__global__ void pack_b_frag(const float* __restrict__ W,
                            _Float16* __restrict__ dst, int K, int N) {
    int total = K * N;
    int ntiles = N >> 4;
    for (int p = blockIdx.x * blockDim.x + threadIdx.x; p < total;
         p += gridDim.x * blockDim.x) {
        int i    = p & 15;
        int lane = (p >> 4) & 31;
        int rest = p >> 9;
        int nt   = rest % ntiles;
        int kt   = rest / ntiles;
        int k = kt * 32 + ((lane >> 4) << 3) + (i < 8 ? i : i + 8);
        int n = nt * 16 + (lane & 15);
        dst[p] = (_Float16)W[(size_t)k * N + n];
    }
}

// ---------------------------------------------------------------------------
// Two-M-panel K-loop shared by both GEMMs. Each wave owns 8 n-tiles and
// 2 M-panels (16 C tiles). B tiles load in groups of 4; each group feeds
// 8 WMMAs (2 panels x 4 tiles), amortizing the load wait 2x.
// ---------------------------------------------------------------------------
template <int KT, int RS>
__device__ inline void gemm_pipeline2(const _Float16* s_a, const v16h* wp,
                                      int NT, int lane,
                                      v8f acc0[8], v8f acc1[8]) {
    v16h bA[4], bB[4];
#pragma unroll
    for (int t = 0; t < 4; ++t) bA[t] = wp[(size_t)t << 5];
#pragma unroll 1
    for (int kt = 0; kt < KT; ++kt) {
        v16h a0 = lds_a_frag(s_a,           RS, kt << 5, lane);  // rows 0-15
        v16h a1 = lds_a_frag(s_a + 16 * RS, RS, kt << 5, lane);  // rows 16-31
        // prefetch group 1 of this step
#pragma unroll
        for (int t = 0; t < 4; ++t)
            bB[t] = wp[(size_t)(kt * NT + 4 + t) << 5];
        // compute group 0: 8 WMMAs per 4 B tiles
#pragma unroll
        for (int t = 0; t < 4; ++t) {
            acc0[t] = wmma_f16(a0, bA[t], acc0[t]);
            acc1[t] = wmma_f16(a1, bA[t], acc1[t]);
        }
        // prefetch group 0 of next step (wrap on last iter: harmless reload)
        int ktn = (kt + 1) & (KT - 1);
#pragma unroll
        for (int t = 0; t < 4; ++t)
            bA[t] = wp[(size_t)(ktn * NT + t) << 5];
        // compute group 1
#pragma unroll
        for (int t = 0; t < 4; ++t) {
            acc0[4 + t] = wmma_f16(a0, bB[t], acc0[4 + t]);
            acc1[4 + t] = wmma_f16(a1, bB[t], acc1[4 + t]);
        }
    }
}

// ---------------------------------------------------------------------------
// GEMM1: h = relu(uf16[B,256] @ W1 + b1) -> h16 [B,1024] f16
// ---------------------------------------------------------------------------
__global__ void __launch_bounds__(256)
gemm1_relu(const _Float16* __restrict__ uf16,
           const v16h*     __restrict__ w1p,
           const float*    __restrict__ b1,
           _Float16*       __restrict__ h16) {
    __shared__ __align__(16) _Float16 s_a[RPB * (IU + PADE)];  // 16.5 KB
    const int lane = threadIdx.x & 31;
    const int wave = threadIdx.x >> 5;
    const int row0 = blockIdx.x * RPB;
    const int NT   = HH >> 4;   // 64

    // Stage A panel: 32 rows x 32 chunks of 16B = 1024 chunks
    for (int c = threadIdx.x; c < RPB * (IU / 8); c += 256) {
        int r = c >> 5, k8 = c & 31;
        async_cp16(&s_a[r * (IU + PADE) + k8 * 8],
                   uf16 + (size_t)(row0 + r) * IU + k8 * 8);
    }
    wait_async0();
    __syncthreads();

    v8f acc0[8] = {}, acc1[8] = {};
    const v16h* wp = w1p + (((size_t)wave * 8) << 5) + lane;
    gemm_pipeline2<(IU >> 5), IU + PADE>(s_a, wp, NT, lane, acc0, acc1);

    // Epilogue: C tile — VGPR r, lane l: M = r + (l>=16 ? 8:0), N = l&15
    const int mrow = (lane >> 4) << 3;
    const int ncol = lane & 15;
#pragma unroll
    for (int t = 0; t < 8; ++t) {
        int n = wave * 128 + t * 16 + ncol;
        float bias = b1[n];
#pragma unroll
        for (int r = 0; r < 8; ++r) {
            float v0 = acc0[t][r] + bias;
            v0 = v0 > 0.0f ? v0 : 0.0f;
            h16[(size_t)(row0 + mrow + r) * HH + n] = (_Float16)v0;
            float v1 = acc1[t][r] + bias;
            v1 = v1 > 0.0f ? v1 : 0.0f;
            h16[(size_t)(row0 + 16 + mrow + r) * HH + n] = (_Float16)v1;
        }
    }
}

// ---------------------------------------------------------------------------
// GEMM2 + fused attention tail (32 rows/block).
// A panel (66 KB) time-aliased with user_emb tile (64 KB) in LDS.
// ---------------------------------------------------------------------------
__global__ void __launch_bounds__(256)
gemm2_attn(const _Float16* __restrict__ h16,
           const v16h*     __restrict__ w2p,
           const float*    __restrict__ b2,
           const int*      __restrict__ item_ids,
           const float*    __restrict__ emb_table,
           const float*    __restrict__ W3,
           const float*    __restrict__ b3,
           float*          __restrict__ out) {
    __shared__ __align__(16) char s_raw[RPB * (ED + PADE) * 2];  // 66 KB
    __shared__ __align__(16) float s_item[RPB * DD];             // 16 KB
    __shared__ float s_dots[RPB * EE];
    __shared__ float s_attn[RPB * EE];
    _Float16* s_a  = (_Float16*)s_raw;   // phase 1: A panel [RPB][ED+PADE]
    _Float16* s_ue = (_Float16*)s_raw;   // phase 2: user_emb [RPB][ED]

    const int lane = threadIdx.x & 31;
    const int wave = threadIdx.x >> 5;
    const int row0 = blockIdx.x * RPB;
    const int NT   = ED >> 4;   // 64

    // Stage A panel: 32 rows x 128 chunks of 16B = 4096 chunks (16/thread)
    for (int c = threadIdx.x; c < RPB * (ED / 8); c += 256) {
        int r = c >> 7, k8 = c & 127;
        async_cp16(&s_a[r * (ED + PADE) + k8 * 8],
                   h16 + (size_t)(row0 + r) * HH + k8 * 8);
    }
    // Stage item embeddings: 32 rows x 32 chunks of 16B = 1024 chunks
    for (int c = threadIdx.x; c < RPB * (DD / 4); c += 256) {
        int r = c >> 5, q = c & 31;
        async_cp16(&s_item[r * DD + q * 4],
                   emb_table + (size_t)item_ids[row0 + r] * DD + q * 4);
    }
    wait_async0();
    __syncthreads();

    v8f acc0[8] = {}, acc1[8] = {};
    const v16h* wp = w2p + (((size_t)wave * 8) << 5) + lane;
    gemm_pipeline2<(HH >> 5), ED + PADE>(s_a, wp, NT, lane, acc0, acc1);

    __syncthreads();   // all waves done reading s_a; safe to alias as s_ue

    const int mrow = (lane >> 4) << 3;
    const int ncol = lane & 15;
#pragma unroll
    for (int t = 0; t < 8; ++t) {
        int n = wave * 128 + t * 16 + ncol;
        float bias = b2[n];
#pragma unroll
        for (int r = 0; r < 8; ++r) {
            s_ue[(mrow + r) * ED + n]        = (_Float16)(acc0[t][r] + bias);
            s_ue[(16 + mrow + r) * ED + n]   = (_Float16)(acc1[t][r] + bias);
        }
    }
    __syncthreads();

    // dots[r][e] = <user_emb[r,e,:], item_emb[r,:]>   (all 256 threads)
    {
        int r = threadIdx.x >> 3, e = threadIdx.x & 7;
        float s = 0.0f;
#pragma unroll 4
        for (int d = 0; d < DD; ++d)
            s += (float)s_ue[r * ED + e * DD + d] * s_item[r * DD + d];
        s_dots[r * EE + e] = s;
    }
    __syncthreads();

    // softmax over E=8 per row (32 threads)
    if (threadIdx.x < RPB) {
        int r = threadIdx.x;
        float mx = s_dots[r * EE];
#pragma unroll
        for (int e = 1; e < EE; ++e) mx = fmaxf(mx, s_dots[r * EE + e]);
        float ex[EE], sum = 0.0f;
#pragma unroll
        for (int e = 0; e < EE; ++e) {
            ex[e] = __expf(s_dots[r * EE + e] - mx);
            sum += ex[e];
        }
        float inv = 1.0f / sum;
#pragma unroll
        for (int e = 0; e < EE; ++e) s_attn[r * EE + e] = ex[e] * inv;
    }
    __syncthreads();

    // out[r][t] = [combined | item_emb | dots] @ W3[:,t] + b3[t]  (256 thr)
    {
        int r = threadIdx.x >> 3, t = threadIdx.x & 7;
        float aw[EE];
#pragma unroll
        for (int e = 0; e < EE; ++e) aw[e] = s_attn[r * EE + e];
        float s = b3[t];
        for (int d = 0; d < DD; ++d) {
            float c = 0.0f;
#pragma unroll
            for (int e = 0; e < EE; ++e)
                c += aw[e] * (float)s_ue[r * ED + e * DD + d];
            s += c * W3[d * TT + t];
            s += s_item[r * DD + d] * W3[(DD + d) * TT + t];
        }
#pragma unroll
        for (int e = 0; e < EE; ++e)
            s += s_dots[r * EE + e] * W3[(2 * DD + e) * TT + t];
        out[(size_t)(row0 + r) * TT + t] = s;
    }
}

// ---------------------------------------------------------------------------
// Launch.  d_in: user_features, item_ids, W1, b1, W2, b2, emb_table, W3, b3
// ---------------------------------------------------------------------------
extern "C" void kernel_launch(void* const* d_in, const int* in_sizes, int n_in,
                              void* d_out, int out_size, void* d_ws, size_t ws_size,
                              hipStream_t stream) {
    const float* uf        = (const float*)d_in[0];
    const int*   item_ids  = (const int*)  d_in[1];
    const float* W1        = (const float*)d_in[2];
    const float* b1        = (const float*)d_in[3];
    const float* W2        = (const float*)d_in[4];
    const float* b2        = (const float*)d_in[5];
    const float* emb_table = (const float*)d_in[6];
    const float* W3        = (const float*)d_in[7];
    const float* b3        = (const float*)d_in[8];
    float* out = (float*)d_out;

    char* ws = (char*)d_ws;
    size_t off = 0;
    auto alloc = [&](size_t bytes) {
        void* p = ws + off;
        off += (bytes + 255) & ~(size_t)255;
        return p;
    };
    _Float16* uf16 = (_Float16*)alloc((size_t)BB * IU * 2);  // 8 MB
    _Float16* h16  = (_Float16*)alloc((size_t)BB * HH * 2);  // 32 MB
    _Float16* w1p  = (_Float16*)alloc((size_t)IU * HH * 2);  // 0.5 MB
    _Float16* w2p  = (_Float16*)alloc((size_t)HH * ED * 2);  // 2 MB
    (void)ws_size; (void)in_sizes; (void)n_in; (void)out_size;

    cvt_f32_to_f16<<<2048, 256, 0, stream>>>(uf, uf16, BB * IU);
    pack_b_frag  <<<512,  256, 0, stream>>>(W1, w1p, IU, HH);
    pack_b_frag  <<<1024, 256, 0, stream>>>(W2, w2p, HH, ED);

    gemm1_relu<<<BB / RPB, 256, 0, stream>>>(uf16, (const v16h*)w1p, b1, h16);
    gemm2_attn<<<BB / RPB, 256, 0, stream>>>(h16, (const v16h*)w2p, b2,
                                             item_ids, emb_table, W3, b3, out);
}
